// TransformerBlock_9440338117179
// MI455X (gfx1250) — matrix operand order repaired and checked by reading, not compiled, not run on hardware
//
#include <hip/hip_runtime.h>
#include <hip/hip_bf16.h>
#include <stdint.h>

// Problem constants (match reference)
#define BB   4
#define SS   1024
#define DD   1024
#define HH   8
#define FF_  4096
#define DK_  128

typedef __attribute__((ext_vector_type(16))) __bf16 v16bf;
typedef __attribute__((ext_vector_type(8)))  float  v8f;

struct bf16frag_raw { uint4 a, b; };          // 32B == v16bf
struct us8          { unsigned short h[8]; }; // 16B == uint4

__device__ __forceinline__ unsigned short f2bf_bits(float f) {
  unsigned u = __builtin_bit_cast(unsigned, f);
  unsigned r = u + 0x7FFFu + ((u >> 16) & 1u);   // round-to-nearest-even
  return (unsigned short)(r >> 16);
}

// Fragment = two 16B contiguous chunks -> two global_load_b128, no repacking.
__device__ __forceinline__ v16bf load_frag(const unsigned short* p0,
                                           const unsigned short* p1) {
  bf16frag_raw raw;
  raw.a = *reinterpret_cast<const uint4*>(p0);
  raw.b = *reinterpret_cast<const uint4*>(p1);
  return __builtin_bit_cast(v16bf, raw);
}

// ---------------------------------------------------------------------------
// bf16 WMMA GEMM, C[z] = act( scale*(A[z] @ B[z]^T) + bias ) + res
// B is stored "output-column major": row n of Bm = K contiguous bf16 that
// multiply into output column n (pre-transposed weights / k / vT).
//   OUTBF  : store bf16 result (else fp32)
//   DOGELU : exact GELU epilogue
//   DORES  : add fp32 residual
//   DOBIAS : add per-column bias
//   CTRANS : store bf16 result transposed (C^T), ldc = #rows of C; packed b128
// Block = 128 thr (4 waves, 2x2), block tile 64x128, wave tile 32x64, K step 32.
// Software-pipelined: next K-slab's 12 b128 fragment loads are issued before
// the current slab's 8 WMMAs, so the WMMA burst hides the load latency.
// Batch z -> (zb, zh) = (z / hdiv, z % hdiv) with independent strides.
// ---------------------------------------------------------------------------
template<bool OUTBF, bool DOGELU, bool DORES, bool DOBIAS, bool CTRANS>
__global__ __launch_bounds__(128) void gemm_bf16_wmma(
    const unsigned short* __restrict__ A,  long long lda, long long sAb, long long sAh,
    const unsigned short* __restrict__ Bm, long long ldb, long long sBb, long long sBh,
    const float* __restrict__ bias,
    const float* __restrict__ res, long long ldres,
    float* __restrict__ Cf, unsigned short* __restrict__ Cb,
    long long ldc, long long sCb, long long sCh,
    int K, int hdiv, float scale) {

  const int tid  = threadIdx.x;
  const int wave = tid >> 5;
  const int lane = tid & 31;

  const int z  = blockIdx.z;
  const int zb = z / hdiv, zh = z % hdiv;
  A  += (long long)zb * sAb + (long long)zh * sAh;
  Bm += (long long)zb * sBb + (long long)zh * sBh;
  const long long cbase = (long long)zb * sCb + (long long)zh * sCh;
  if constexpr (OUTBF) Cb += cbase; else Cf += cbase;

  const int mb = blockIdx.y * 64  + (wave >> 1) * 32;   // wave's 32 output rows
  const int nb = blockIdx.x * 128 + (wave & 1) * 64;    // wave's 64 output cols
  const int mrow = lane & 15;                           // A row / B row in tile
  const int kh   = lane >> 4;                           // lane-group K select

  // Hoisted row base pointers (ISA 16-bit A 16x32 layout: elems 0..7 at K
  // kh*8.., elems 8..15 at K 16+kh*8..; B fragment: same K layout, col n)
  const unsigned short* Ar0 = A + (long long)(mb + mrow) * lda;
  const unsigned short* Ar1 = Ar0 + 16 * lda;
  const unsigned short* Br0 = Bm + (long long)(nb + mrow) * ldb;
  const unsigned short* Br1 = Br0 + 16 * ldb;
  const unsigned short* Br2 = Br0 + 32 * ldb;
  const unsigned short* Br3 = Br0 + 48 * ldb;

  v8f acc[2][4];
#pragma unroll
  for (int mt = 0; mt < 2; ++mt)
#pragma unroll
    for (int t = 0; t < 4; ++t)
#pragma unroll
      for (int r = 0; r < 8; ++r) acc[mt][t][r] = 0.0f;

  const int kaoff = kh * 8;    // A fragment K base for this lane group
  const int kboff = kh * 8;    // B fragment K base for this lane group

  // Prologue: load first K-slab's fragments.
  v16bf a0 = load_frag(Ar0 + kaoff, Ar0 + kaoff + 16);
  v16bf a1 = load_frag(Ar1 + kaoff, Ar1 + kaoff + 16);
  v16bf b0 = load_frag(Br0 + kboff, Br0 + kboff + 16);
  v16bf b1 = load_frag(Br1 + kboff, Br1 + kboff + 16);
  v16bf b2 = load_frag(Br2 + kboff, Br2 + kboff + 16);
  v16bf b3 = load_frag(Br3 + kboff, Br3 + kboff + 16);

  // Steady state: issue next slab's loads, then WMMA on the current slab.
  for (int kk = 0; kk + 32 < K; kk += 32) {
    const int ka = kk + 32 + kaoff;
    const int kb = kk + 32 + kboff;
    const v16bf na0 = load_frag(Ar0 + ka, Ar0 + ka + 16);
    const v16bf na1 = load_frag(Ar1 + ka, Ar1 + ka + 16);
    const v16bf nb0 = load_frag(Br0 + kb, Br0 + kb + 16);
    const v16bf nb1 = load_frag(Br1 + kb, Br1 + kb + 16);
    const v16bf nb2 = load_frag(Br2 + kb, Br2 + kb + 16);
    const v16bf nb3 = load_frag(Br3 + kb, Br3 + kb + 16);
    // Speculative prefetch one more slab ahead (silently dropped if OOB).
    __builtin_prefetch(Ar0 + ka + 32, 0, 3);             // global_prefetch_b8
    __builtin_prefetch(Ar1 + ka + 32, 0, 3);

    acc[0][0] = __builtin_amdgcn_wmma_f32_16x16x32_bf16(false, a0, false, b0, (short)0, acc[0][0], false, false);
    acc[1][0] = __builtin_amdgcn_wmma_f32_16x16x32_bf16(false, a1, false, b0, (short)0, acc[1][0], false, false);
    acc[0][1] = __builtin_amdgcn_wmma_f32_16x16x32_bf16(false, a0, false, b1, (short)0, acc[0][1], false, false);
    acc[1][1] = __builtin_amdgcn_wmma_f32_16x16x32_bf16(false, a1, false, b1, (short)0, acc[1][1], false, false);
    acc[0][2] = __builtin_amdgcn_wmma_f32_16x16x32_bf16(false, a0, false, b2, (short)0, acc[0][2], false, false);
    acc[1][2] = __builtin_amdgcn_wmma_f32_16x16x32_bf16(false, a1, false, b2, (short)0, acc[1][2], false, false);
    acc[0][3] = __builtin_amdgcn_wmma_f32_16x16x32_bf16(false, a0, false, b3, (short)0, acc[0][3], false, false);
    acc[1][3] = __builtin_amdgcn_wmma_f32_16x16x32_bf16(false, a1, false, b3, (short)0, acc[1][3], false, false);

    a0 = na0; a1 = na1; b0 = nb0; b1 = nb1; b2 = nb2; b3 = nb3;
  }

  // Epilogue slab.
  acc[0][0] = __builtin_amdgcn_wmma_f32_16x16x32_bf16(false, a0, false, b0, (short)0, acc[0][0], false, false);
  acc[1][0] = __builtin_amdgcn_wmma_f32_16x16x32_bf16(false, a1, false, b0, (short)0, acc[1][0], false, false);
  acc[0][1] = __builtin_amdgcn_wmma_f32_16x16x32_bf16(false, a0, false, b1, (short)0, acc[0][1], false, false);
  acc[1][1] = __builtin_amdgcn_wmma_f32_16x16x32_bf16(false, a1, false, b1, (short)0, acc[1][1], false, false);
  acc[0][2] = __builtin_amdgcn_wmma_f32_16x16x32_bf16(false, a0, false, b2, (short)0, acc[0][2], false, false);
  acc[1][2] = __builtin_amdgcn_wmma_f32_16x16x32_bf16(false, a1, false, b2, (short)0, acc[1][2], false, false);
  acc[0][3] = __builtin_amdgcn_wmma_f32_16x16x32_bf16(false, a0, false, b3, (short)0, acc[0][3], false, false);
  acc[1][3] = __builtin_amdgcn_wmma_f32_16x16x32_bf16(false, a1, false, b3, (short)0, acc[1][3], false, false);

  // Store. C/D layout: VGPR r -> row r (lanes 0-15) / row 8+r (lanes 16-31),
  // N = lane&15. For CTRANS the 8 rows of one column are contiguous -> b128.
  const int row0 = mb + (lane >> 4) * 8;
  const int coll = lane & 15;
#pragma unroll
  for (int mt = 0; mt < 2; ++mt) {
    const int rbase = row0 + mt * 16;
#pragma unroll
    for (int t = 0; t < 4; ++t) {
      const int col = nb + t * 16 + coll;
      float vals[8];
#pragma unroll
      for (int r = 0; r < 8; ++r) {
        float v = acc[mt][t][r] * scale;
        if constexpr (DOBIAS) v += bias[col];
        if constexpr (DOGELU) v = 0.5f * v * (1.0f + erff(v * 0.70710678118654752f));
        if constexpr (DORES)  v += res[(long long)(rbase + r) * ldres + col];
        vals[r] = v;
      }
      if constexpr (CTRANS) {
        us8 o;
#pragma unroll
        for (int r = 0; r < 8; ++r) o.h[r] = f2bf_bits(vals[r]);
        *reinterpret_cast<uint4*>(&Cb[(long long)col * ldc + rbase]) =
            __builtin_bit_cast(uint4, o);
      } else if constexpr (OUTBF) {
#pragma unroll
        for (int r = 0; r < 8; ++r)
          Cb[(long long)(rbase + r) * ldc + col] = f2bf_bits(vals[r]);
      } else {
#pragma unroll
        for (int r = 0; r < 8; ++r)
          Cf[(long long)(rbase + r) * ldc + col] = vals[r];
      }
    }
  }
}

// ---------------------------------------------------------------------------
// LayerNorm over last dim (D=1024), one block per row, emit bf16 for WMMA.
// ---------------------------------------------------------------------------
__global__ __launch_bounds__(256) void layernorm_to_bf16(
    const float* __restrict__ x, const float* __restrict__ g,
    const float* __restrict__ b, unsigned short* __restrict__ out) {
  const long long row = blockIdx.x;
  const float* xr = x + row * (long long)DD;
  __shared__ float red[256];
  const int tid = threadIdx.x;

  float s = 0.0f;
  for (int i = tid; i < DD; i += 256) s += xr[i];
  red[tid] = s; __syncthreads();
  for (int o = 128; o > 0; o >>= 1) { if (tid < o) red[tid] += red[tid + o]; __syncthreads(); }
  const float mean = red[0] * (1.0f / DD);
  __syncthreads();

  float vs = 0.0f;
  for (int i = tid; i < DD; i += 256) { const float d = xr[i] - mean; vs += d * d; }
  red[tid] = vs; __syncthreads();
  for (int o = 128; o > 0; o >>= 1) { if (tid < o) red[tid] += red[tid + o]; __syncthreads(); }
  const float rstd = rsqrtf(red[0] * (1.0f / DD) + 1e-5f);

  unsigned short* orow = out + row * (long long)DD;
  for (int i = tid; i < DD; i += 256)
    orow[i] = f2bf_bits((xr[i] - mean) * rstd * g[i] + b[i]);
}

// Row softmax over S=1024, emit bf16 probabilities.
__global__ __launch_bounds__(256) void softmax_to_bf16(
    const float* __restrict__ sc, unsigned short* __restrict__ p) {
  const long long row = blockIdx.x;
  const float* r = sc + row * (long long)SS;
  __shared__ float red[256];
  const int tid = threadIdx.x;

  float m = -3.402823466e38f;
  for (int i = tid; i < SS; i += 256) m = fmaxf(m, r[i]);
  red[tid] = m; __syncthreads();
  for (int o = 128; o > 0; o >>= 1) { if (tid < o) red[tid] = fmaxf(red[tid], red[tid + o]); __syncthreads(); }
  const float M = red[0];
  __syncthreads();

  float s = 0.0f;
  for (int i = tid; i < SS; i += 256) s += __expf(r[i] - M);
  red[tid] = s; __syncthreads();
  for (int o = 128; o > 0; o >>= 1) { if (tid < o) red[tid] += red[tid + o]; __syncthreads(); }
  const float inv = 1.0f / red[0];

  unsigned short* pr = p + row * (long long)SS;
  for (int i = tid; i < SS; i += 256) pr[i] = f2bf_bits(__expf(r[i] - M) * inv);
}

// Transpose-convert a [K,N] fp32 weight to [N,K] bf16 (read coalesced).
__global__ void f32_to_bf16_transpose(const float* __restrict__ in,
                                      unsigned short* __restrict__ out,
                                      int Kdim, int Ndim) {
  const long long i = (long long)blockIdx.x * blockDim.x + threadIdx.x;
  if (i >= (long long)Kdim * Ndim) return;
  const int n = (int)(i % Ndim);
  const int k = (int)(i / Ndim);
  out[(long long)n * Kdim + k] = f2bf_bits(in[i]);
}

// Reference's raw ctx.reshape(S,B,D,H).sum(-1) head-sum + residual:
// x1[b,s,d] = x[b,s,d] + sum_h ctx_flat[((s*B+b)*D+d)*H + h]
__global__ void attn_combine_residual(const float* __restrict__ x,
                                      const float* __restrict__ ctx,
                                      float* __restrict__ x1) {
  const long long idx = (long long)blockIdx.x * blockDim.x + threadIdx.x;
  if (idx >= (long long)BB * SS * DD) return;
  const int d = (int)(idx % DD);
  const long long t = idx / DD;
  const int s = (int)(t % SS);
  const int b = (int)(t / SS);
  const long long base = (((long long)s * BB + b) * DD + d) * HH;
  float a = 0.0f;
#pragma unroll
  for (int h = 0; h < HH; ++h) a += ctx[base + h];
  x1[idx] = x[idx] + a;
}

// ---------------------------------------------------------------------------
extern "C" void kernel_launch(void* const* d_in, const int* in_sizes, int n_in,
                              void* d_out, int out_size, void* d_ws, size_t ws_size,
                              hipStream_t stream) {
  (void)in_sizes; (void)n_in; (void)out_size; (void)ws_size;

  const float* x   = (const float*)d_in[0];
  /* d_in[1] = mask (unused in infer path) */
  const float* g1  = (const float*)d_in[2];
  const float* b1  = (const float*)d_in[3];
  const float* Wq  = (const float*)d_in[4];
  const float* bq  = (const float*)d_in[5];
  const float* Wk  = (const float*)d_in[6];
  const float* bk  = (const float*)d_in[7];
  const float* Wv  = (const float*)d_in[8];
  const float* bv  = (const float*)d_in[9];
  const float* g2  = (const float*)d_in[10];
  const float* b2  = (const float*)d_in[11];
  const float* W1  = (const float*)d_in[12];
  const float* bw1 = (const float*)d_in[13];
  const float* W2  = (const float*)d_in[14];
  const float* bw2 = (const float*)d_in[15];
  float* out = (float*)d_out;

  // Bump allocator over d_ws (256B aligned regions)
  char* wsb = (char*)d_ws;
  size_t off = 0;
  auto alloc = [&](size_t bytes) -> void* {
    void* p = wsb + off;
    off = (off + bytes + 255) & ~(size_t)255;
    return p;
  };
  const long long M = (long long)BB * SS;          // 4096 activation rows
  unsigned short* xn_bf  = (unsigned short*)alloc(M * DD * 2);
  unsigned short* wqT    = (unsigned short*)alloc((long long)DD * DD * 2);       // [D,D]
  unsigned short* wkT    = (unsigned short*)alloc((long long)DD * DD * 2);       // [D,D]
  unsigned short* wvT    = (unsigned short*)alloc((long long)HH * DD * DD * 2);  // [H*D,D]
  unsigned short* w1T    = (unsigned short*)alloc((long long)FF_ * DD * 2);      // [FF,D]
  unsigned short* w2T    = (unsigned short*)alloc((long long)DD * FF_ * 2);      // [D,FF]
  unsigned short* q_bf   = (unsigned short*)alloc(M * DD * 2);                   // [B*S,D]
  unsigned short* k_bf   = (unsigned short*)alloc(M * DD * 2);                   // [B*S,D]
  unsigned short* vT_bf  = (unsigned short*)alloc((long long)HH * DD * M * 2);   // [H*D,B*S]
  float*          sc_f   = (float*)alloc((long long)BB * HH * SS * SS * 4);
  unsigned short* p_bf   = (unsigned short*)alloc((long long)BB * HH * SS * SS * 2);
  float*          ctx_f  = (float*)alloc((long long)BB * HH * SS * DD * 4);
  float*          x1_f   = (float*)alloc(M * DD * 4);
  unsigned short* xn2_bf = (unsigned short*)alloc(M * DD * 2);
  unsigned short* t1_bf  = (unsigned short*)alloc(M * FF_ * 2);

  // 1) LN1 -> bf16
  layernorm_to_bf16<<<(int)M, 256, 0, stream>>>(x, g1, b1, xn_bf);

  // 2) Weights -> bf16, pre-transposed to output-column-major for the B^T path
  auto cvtT = [&](const float* src, unsigned short* dst, int Kd, int Nd) {
    const long long n = (long long)Kd * Nd;
    f32_to_bf16_transpose<<<(int)((n + 255) / 256), 256, 0, stream>>>(src, dst, Kd, Nd);
  };
  cvtT(Wq, wqT, DD, DD);
  cvtT(Wk, wkT, DD, DD);
  cvtT(Wv, wvT, DD, HH * DD);
  cvtT(W1, w1T, DD, FF_);
  cvtT(W2, w2T, FF_, DD);

  // 3) q = xn@Wq+bq, k = xn@Wk+bk (bf16 row-major); v = xn@Wv+bv stored
  //    TRANSPOSED (vT: [H*D, B*S]) straight from the epilogue (packed b128).
  gemm_bf16_wmma<true, false, false, true, false><<<dim3(DD / 128, M / 64, 1), 128, 0, stream>>>(
      xn_bf, DD, 0, 0, wqT, DD, 0, 0, bq, nullptr, 0,
      nullptr, q_bf, DD, 0, 0, DD, 1, 1.0f);
  gemm_bf16_wmma<true, false, false, true, false><<<dim3(DD / 128, M / 64, 1), 128, 0, stream>>>(
      xn_bf, DD, 0, 0, wkT, DD, 0, 0, bk, nullptr, 0,
      nullptr, k_bf, DD, 0, 0, DD, 1, 1.0f);
  gemm_bf16_wmma<true, false, false, true, true><<<dim3((HH * DD) / 128, M / 64, 1), 128, 0, stream>>>(
      xn_bf, DD, 0, 0, wvT, DD, 0, 0, bv, nullptr, 0,
      nullptr, vT_bf, M /*ldc of C^T*/, 0, 0, DD, 1, 1.0f);

  // 4) scores[b,h] = q_h @ k_h^T / sqrt(DK)   (z = b*H + h; k row-major IS B^T)
  gemm_bf16_wmma<false, false, false, false, false><<<dim3(SS / 128, SS / 64, BB * HH), 128, 0, stream>>>(
      q_bf, DD, (long long)SS * DD, DK_,
      k_bf, DD, (long long)SS * DD, DK_,
      nullptr, nullptr, 0,
      sc_f, nullptr, SS, (long long)HH * SS * SS, (long long)SS * SS,
      DK_, HH, 0.08838834764831845f);

  // 5) softmax rows -> bf16 p
  softmax_to_bf16<<<BB * HH * SS, 256, 0, stream>>>(sc_f, p_bf);

  // 6) ctx[b,h] = p[b,h] @ v[b,h]; v supplied via vT (rows h*D+n, cols b*S+k)
  gemm_bf16_wmma<false, false, false, false, false><<<dim3(DD / 128, SS / 64, BB * HH), 128, 0, stream>>>(
      p_bf, SS, (long long)HH * SS * SS, (long long)SS * SS,
      vT_bf, M, SS /*batch b -> col offset b*S*/, (long long)DD * M /*head h -> row offset h*D*/,
      nullptr, nullptr, 0,
      ctx_f, nullptr, DD, (long long)HH * SS * DD, (long long)SS * DD,
      SS, HH, 1.0f);

  // 7) head-sum (raw reshape semantics) + residual -> x1
  attn_combine_residual<<<(int)((M * DD + 255) / 256), 256, 0, stream>>>(x, ctx_f, x1_f);

  // 8) LN2 -> bf16
  layernorm_to_bf16<<<(int)M, 256, 0, stream>>>(x1_f, g2, b2, xn2_bf);

  // 9) t1 = gelu(xn2 @ W1 + bw1)  (bf16 out)
  gemm_bf16_wmma<true, true, false, true, false><<<dim3(FF_ / 128, M / 64, 1), 128, 0, stream>>>(
      xn2_bf, DD, 0, 0, w1T, DD, 0, 0, bw1, nullptr, 0,
      nullptr, t1_bf, FF_, 0, 0, DD, 1, 1.0f);

  // 10) out = x1 + (t1 @ W2 + bw2)  (fp32 out, fused residual)
  gemm_bf16_wmma<false, false, true, true, false><<<dim3(DD / 128, M / 64, 1), 128, 0, stream>>>(
      t1_bf, FF_, 0, 0, w2T, FF_, 0, 0, bw2, x1_f, DD,
      out, nullptr, DD, 0, 0, FF_, 1, 1.0f);
}
